// LSTMCell_55697135894882
// MI455X (gfx1250) — matrix operand order repaired and checked by reading, not compiled, not run on hardware
//
#include <hip/hip_runtime.h>

typedef __attribute__((ext_vector_type(16))) __bf16 v16bf;
typedef __attribute__((ext_vector_type(8)))  float  v8f;

#define HID    128
#define KDIM   256   // 2*H
#define NDIM   512   // 4*H
#define MTILE  64
#define THREADS 512

// ---------------------------------------------------------------------------
// Kernel 1: split w[256][512] (f32, row-major) into bf16 hi/lo, stored
// column-major wt[col][k] so B fragments are two 16-byte runs per lane.
// ---------------------------------------------------------------------------
__global__ void lstm_wconv(const float* __restrict__ w,
                           __bf16* __restrict__ wt_hi,
                           __bf16* __restrict__ wt_lo) {
  int idx = blockIdx.x * blockDim.x + threadIdx.x;   // 0 .. 256*512-1
  int k   = idx >> 9;            // row of w
  int col = idx & (NDIM - 1);    // col of w
  float v = w[idx];
  __bf16 hi = (__bf16)v;
  wt_hi[col * KDIM + k] = hi;
  wt_lo[col * KDIM + k] = (__bf16)(v - (float)hi);
}

__device__ __forceinline__ v16bf ld_frag16(const __bf16* p0, const __bf16* p1) {
  v16bf r;
  ((float4*)&r)[0] = *(const float4*)p0;   // elements 0..7
  ((float4*)&r)[1] = *(const float4*)p1;   // elements 8..15
  return r;
}

__device__ __forceinline__ float fast_sigmoid(float x) {
  return 1.0f / (1.0f + __expf(-x));
}
__device__ __forceinline__ float fast_tanh(float x) {
  // 1 - 2/(e^{2x}+1): correct saturation at +-1 for |x| large
  return 1.0f - 2.0f / (__expf(2.0f * x) + 1.0f);
}

// ---------------------------------------------------------------------------
// Kernel 2: 64 rows/block. Wave (rg,u) owns 16 rows x 32 units across all 4
// gates -> 8 accumulator tiles; gate recombination is register-local.
// ---------------------------------------------------------------------------
__global__ __launch_bounds__(THREADS, 1)
void lstm_main(const float* __restrict__ x, const float* __restrict__ h,
               const float* __restrict__ c,
               const __bf16* __restrict__ wt_hi, const __bf16* __restrict__ wt_lo,
               float* __restrict__ out, int batch) {
  extern __shared__ __bf16 smem[];
  __bf16* a_hi = smem;                   // [MTILE][KDIM] bf16 hi of concat(x,h)
  __bf16* a_lo = smem + MTILE * KDIM;    // [MTILE][KDIM] bf16 lo residual

  const int rb = blockIdx.x * MTILE;

  // Stage A = concat(x,h) rows into LDS as bf16 hi/lo (coalesced reads).
  for (int e = threadIdx.x; e < MTILE * KDIM; e += THREADS) {
    int row = e >> 8;
    int k   = e & (KDIM - 1);
    float v = (k < HID) ? x[(size_t)(rb + row) * HID + k]
                        : h[(size_t)(rb + row) * HID + (k - HID)];
    __bf16 hi = (__bf16)v;
    a_hi[e] = hi;
    a_lo[e] = (__bf16)(v - (float)hi);
  }
  __syncthreads();

  const int lane  = threadIdx.x & 31;
  const int wid   = threadIdx.x >> 5;   // 0..15
  const int rg    = wid >> 2;           // row group 0..3 (16 rows each)
  const int u0    = (wid & 3) * 32;     // unit offset 0/32/64/96
  const int n16   = lane & 15;
  const int half  = lane >> 4;          // lane 16..31 -> 1
  const int abase = half << 3;          // A frag K base: 0 or 8
  const int klo   = half << 3;          // B frag K base: 0 or 8

  v8f acc[8];
  #pragma unroll
  for (int t = 0; t < 8; ++t)
    #pragma unroll
    for (int e2 = 0; e2 < 8; ++e2) acc[t][e2] = 0.0f;

  // A fragment source row in LDS (A 16x32 bf16 layout: lane holds row M=lane%16,
  // runs K = base+0..7 and K = base+16..23, base = (lane<16)?0:8)
  const __bf16* arow_hi = a_hi + (rg * 16 + n16) * KDIM;
  const __bf16* arow_lo = a_lo + (rg * 16 + n16) * KDIM;

  for (int kk = 0; kk < KDIM; kk += 32) {
    v16bf ah = ld_frag16(arow_hi + kk + abase, arow_hi + kk + abase + 16);
    v16bf al = ld_frag16(arow_lo + kk + abase, arow_lo + kk + abase + 16);
    #pragma unroll
    for (int gi = 0; gi < 4; ++gi) {       // gates i,f,g,o
      #pragma unroll
      for (int s = 0; s < 2; ++s) {        // two 16-col subtiles per gate
        const int cb = gi * HID + u0 + s * 16;
        // B 32x16 bf16 layout: lane holds col N=lane%16, K = klo+0..7 and klo+16..23
        const __bf16* bp_hi = wt_hi + (size_t)(cb + n16) * KDIM + kk + klo;
        const __bf16* bp_lo = wt_lo + (size_t)(cb + n16) * KDIM + kk + klo;
        v16bf bh = ld_frag16(bp_hi, bp_hi + 16);
        v16bf bl = ld_frag16(bp_lo, bp_lo + 16);
        const int t = gi * 2 + s;
        acc[t] = __builtin_amdgcn_wmma_f32_16x16x32_bf16(
            false, ah, false, bh, (short)0, acc[t], false, false);
        acc[t] = __builtin_amdgcn_wmma_f32_16x16x32_bf16(
            false, al, false, bh, (short)0, acc[t], false, false);
        acc[t] = __builtin_amdgcn_wmma_f32_16x16x32_bf16(
            false, ah, false, bl, (short)0, acc[t], false, false);
      }
    }
  }

  // Epilogue: gates + state update, fully register-local per wave.
  // C/D layout: VGPR v -> row = v + half*8, col = n16.
  float* hout = out;
  float* cout = out + (size_t)batch * HID;
  #pragma unroll
  for (int s = 0; s < 2; ++s) {
    #pragma unroll
    for (int v = 0; v < 8; ++v) {
      const int row  = rb + rg * 16 + v + half * 8;
      const int unit = u0 + s * 16 + n16;
      float iv = fast_sigmoid(acc[0 * 2 + s][v]);
      float fv = fast_sigmoid(acc[1 * 2 + s][v]);
      float gv = fast_tanh   (acc[2 * 2 + s][v]);
      float ov = fast_sigmoid(acc[3 * 2 + s][v]);
      float cold = c[(size_t)row * HID + unit];
      float cnew = fv * cold + iv * gv;
      float hnew = ov * fast_tanh(cnew);
      hout[(size_t)row * HID + unit] = hnew;
      cout[(size_t)row * HID + unit] = cnew;
    }
  }
}

extern "C" void kernel_launch(void* const* d_in, const int* in_sizes, int n_in,
                              void* d_out, int out_size, void* d_ws, size_t ws_size,
                              hipStream_t stream) {
  const float* x = (const float*)d_in[0];
  const float* h = (const float*)d_in[1];
  const float* c = (const float*)d_in[2];
  const float* w = (const float*)d_in[3];
  const int batch = in_sizes[0] / HID;   // 131072

  __bf16* wt_hi = (__bf16*)d_ws;                 // 256 KB
  __bf16* wt_lo = wt_hi + (size_t)KDIM * NDIM;   // 256 KB

  lstm_wconv<<<(KDIM * NDIM) / THREADS, THREADS, 0, stream>>>(w, wt_hi, wt_lo);

  const int blocks = batch / MTILE;                       // 2048
  const size_t lds = (size_t)MTILE * KDIM * 2 * sizeof(__bf16);  // 64 KB
  lstm_main<<<blocks, THREADS, lds, stream>>>(x, h, c, wt_hi, wt_lo,
                                              (float*)d_out, batch);
}
